// QuantLinear_46377056862601
// MI455X (gfx1250) — 8-bit matrix operand order repaired and checked by reading, compiled, not run on hardware
//
#include <hip/hip_runtime.h>
#include <hip/hip_bf16.h>


typedef int v8i __attribute__((ext_vector_type(8)));

#define W_SCALE 0.02f
#define A_SCALE 0.05f

__device__ __forceinline__ int quant1(float v, float inv_scale) {
    float r = rintf(v * inv_scale);
    r = fminf(fmaxf(r, -128.0f), 127.0f);
    return (int)r;
}

__global__ void quant_f32_to_i8_kernel(const float* __restrict__ in,
                                       int* __restrict__ out_packed,
                                       int n4, float inv_scale) {
    int i = blockIdx.x * blockDim.x + threadIdx.x;
    if (i >= n4) return;
    float4 v = ((const float4*)in)[i];
    int a = quant1(v.x, inv_scale);
    int b = quant1(v.y, inv_scale);
    int c = quant1(v.z, inv_scale);
    int d = quant1(v.w, inv_scale);
    out_packed[i] = (a & 255) | ((b & 255) << 8) | ((c & 255) << 16) | ((d & 255) << 24);
}

__global__ __launch_bounds__(256) void wmma_i8_gemm_kernel(
    const signed char* __restrict__ Aq,
    const signed char* __restrict__ Bq,
    const float* __restrict__ bias,
    float* __restrict__ out,
    int N, int O, int E) {

    const int lane  = threadIdx.x & 31;
    const int wave  = threadIdx.x >> 5;
    const int waveM = wave >> 2;
    const int waveN = wave & 3;
    const int r     = lane & 15;
    const int hi    = lane >> 4;

    const int mBase = blockIdx.y * 128 + waveM * 64;
    const int nBase = blockIdx.x * 128 + waveN * 32;

    const signed char* aRow[4];
#pragma unroll
    for (int m = 0; m < 4; ++m)
        aRow[m] = Aq + (size_t)(mBase + m * 16 + r) * E + hi * 16;

    const signed char* bRow[2];
#pragma unroll
    for (int n = 0; n < 2; ++n)
        bRow[n] = Bq + (size_t)(nBase + n * 16 + r) * E + hi * 16;

    v8i acc[4][2];
#pragma unroll
    for (int m = 0; m < 4; ++m)
#pragma unroll
        for (int n = 0; n < 2; ++n) {
            v8i z = {0, 0, 0, 0, 0, 0, 0, 0};
            acc[m][n] = z;
        }

    for (int k0 = 0; k0 < E; k0 += 64) {
        __builtin_prefetch(aRow[0] + k0 + 64, 0, 3);
        __builtin_prefetch(bRow[0] + k0 + 64, 0, 3);

        v8i a[4];
#pragma unroll
        for (int m = 0; m < 4; ++m) {
#pragma unroll
            for (int j = 0; j < 4; ++j) {
                int2 t = *(const int2*)(aRow[m] + k0 + 32 * (j >> 1) + 8 * (j & 1));
                a[m][2 * j]     = t.x;
                a[m][2 * j + 1] = t.y;
            }
        }

        v8i b[2];
#pragma unroll
        for (int n = 0; n < 2; ++n) {
            int4 t0 = *(const int4*)(bRow[n] + k0);
            int4 t1 = *(const int4*)(bRow[n] + k0 + 32);
            b[n][0] = t0.x; b[n][1] = t0.y; b[n][2] = t0.z; b[n][3] = t0.w;
            b[n][4] = t1.x; b[n][5] = t1.y; b[n][6] = t1.z; b[n][7] = t1.w;
        }

#pragma unroll
        for (int m = 0; m < 4; ++m)
#pragma unroll
            for (int n = 0; n < 2; ++n)
                acc[m][n] = __builtin_amdgcn_wmma_i32_16x16x64_iu8(
                     true, a[m],  true, b[n],
                    acc[m][n],  false,  false);
    }

    const float scale = W_SCALE * A_SCALE;
#pragma unroll
    for (int n = 0; n < 2; ++n) {
        const int col = nBase + n * 16 + r;
        const float bs = bias[col];
#pragma unroll
        for (int m = 0; m < 4; ++m) {
            const int rowBase = mBase + m * 16 + 8 * hi;
#pragma unroll
            for (int g = 0; g < 8; ++g) {
                out[(size_t)(rowBase + g) * O + col] =
                    (float)acc[m][n][g] * scale + bs;
            }
        }
    }
}

extern "C" void kernel_launch(void* const* d_in, const int* in_sizes, int n_in,
                              void* d_out, int out_size, void* d_ws, size_t ws_size,
                              hipStream_t stream) {
    const float* x    = (const float*)d_in[0];
    const float* w    = (const float*)d_in[1];
    const float* bias = (const float*)d_in[2];

    const int O = in_sizes[2];
    const int E = in_sizes[1] / O;
    const int N = in_sizes[0] / E;

    signed char* xq = (signed char*)d_ws;
    signed char* wq = (signed char*)d_ws + (size_t)N * E;

    {
        int n4 = (N * E) / 4;
        int blocks = (n4 + 255) / 256;
        quant_f32_to_i8_kernel<<<blocks, 256, 0, stream>>>(
            x, (int*)xq, n4, 1.0f / A_SCALE);
    }
    {
        int n4 = (O * E) / 4;
        int blocks = (n4 + 255) / 256;
        quant_f32_to_i8_kernel<<<blocks, 256, 0, stream>>>(
            w, (int*)wq, n4, 1.0f / W_SCALE);
    }

    {
        dim3 grid(O / 128, N / 128);
        dim3 block(256);
        wmma_i8_gemm_kernel<<<grid, block, 0, stream>>>(
            xq, wq, bias, (float*)d_out, N, O, E);
    }
}
